// Qwen3Attention_48275432407040
// MI455X (gfx1250) — compile-verified, matrix operand order checked by reading, not run on hardware
//
#include <hip/hip_runtime.h>

// ---------------------------------------------------------------------------
// Qwen3 attention block for MI455X (gfx1250, wave32, WMMA bf16 16x16x32,
// TDM tensor_load_to_lds staging for the projection GEMMs).
// ---------------------------------------------------------------------------

#define T_TOK 2048
#define HIDN  2048
#define NH    16
#define NKV   8
#define HD    128
#define QKV_N ((NH + 2 * NKV) * HD)   // 4096

#define KSTEP 32                      // K slab staged per TDM tile
#define PITCH 40                      // 32 elems + 8 pad elems (TDM pad-on-load)

typedef __attribute__((ext_vector_type(16))) __bf16 bf16x16;
typedef __attribute__((ext_vector_type(8)))  __bf16 bf16x8;
typedef __attribute__((ext_vector_type(4)))  __bf16 bf16x4;
typedef __attribute__((ext_vector_type(8)))  float  f32x8;
typedef __attribute__((ext_vector_type(4)))  unsigned int u32x4;
typedef __attribute__((ext_vector_type(8)))  int i32x8;
typedef __attribute__((ext_vector_type(4)))  int i32x4;

// Load a 16-element bf16 A/B fragment as two contiguous 16-byte chunks.
// Matches CDNA5 16-bit matrix layout: lanes 0-15 hold K=[0..7]+[16..23],
// lanes 16-31 hold K=[8..15]+[24..31] (caller passes half-adjusted offsets).
__device__ __forceinline__ bf16x16 load_frag(const __bf16* base, int o0, int o1) {
  union { bf16x16 v; bf16x8 h[2]; } u;
  u.h[0] = *(const bf16x8*)(base + o0);
  u.h[1] = *(const bf16x8*)(base + o1);
  return u.v;
}

__device__ __forceinline__ f32x8 wmma_bf16(bf16x16 a, bf16x16 b, f32x8 c) {
  return __builtin_amdgcn_wmma_f32_16x16x32_bf16(false, a, false, b, (short)0, c,
                                                 false, false);
}

// Generic pointer to a __shared__ object: low 32 bits are the LDS byte offset
// (flat LDS aperture truncation, ISA 10.2).
__device__ __forceinline__ unsigned int lds_off(const void* p) {
  return (unsigned int)(uintptr_t)p;
}

// Issue a TDM 2-D tile load: tile = KSTEP x 128 bf16 elements out of a
// row-major tensor with row stride `strideElems`, padded on LDS store by
// 16B per 64B row (pitch 80B = PITCH elems). D# packing per CDNA5 ISA 8.3/8.4.
// This toolchain's builtin takes 6 args: (g0, g1, g2, g3, g_extra, cpol).
__device__ __forceinline__ void tdm_load_tile(unsigned int ldsAddr,
                                              const void* gptr,
                                              unsigned int strideElems,
                                              unsigned int tdim0,
                                              unsigned int tdim1) {
  unsigned long long ga = (unsigned long long)(uintptr_t)gptr;
  u32x4 g0;
  g0.x = 1u;                                           // count=1 (valid), user mode
  g0.y = ldsAddr;                                      // lds_addr -> bits[63:32]
  g0.z = (unsigned int)ga;                             // global_addr[31:0]
  g0.w = (unsigned int)((ga >> 32) & 0x01ffffffu) | (2u << 30);  // addr hi + type=2
  i32x8 g1;
  // wg_mask=0 | data_size=1 (2B) | pad_enable | pad_interval=3 (16 DW = 64B row)
  // | pad_amount=3 (4 DW = 16B)
  g1[0] = (1 << 16) | (1 << 20) | (3 << 22) | (3 << 25);
  g1[1] = (int)((tdim0 & 0xffffu) << 16);                       // tensor_dim0 lo
  g1[2] = (int)((tdim0 >> 16) | ((tdim1 & 0xffffu) << 16));     // dim0 hi | dim1 lo
  g1[3] = (int)((tdim1 >> 16) | ((unsigned int)KSTEP << 16));   // dim1 hi | tile_dim0
  g1[4] = 128;                                                  // tile_dim1=128, tile_dim2=0
  g1[5] = (int)strideElems;                                     // tensor_dim0_stride lo32
  g1[6] = 0;                                                    // stride0 hi | stride1 lo
  g1[7] = 0;                                                    // stride1 hi
  i32x4 z4 = {0, 0, 0, 0};
  i32x8 z8 = {0, 0, 0, 0, 0, 0, 0, 0};
  __builtin_amdgcn_tensor_load_to_lds(g0, g1, z4, z4, z8, 0);
}

// ---------------------------------------------------------------------------
// fp32 -> bf16 elementwise convert (4 elems / thread, n divisible by 4)
// ---------------------------------------------------------------------------
__global__ void __launch_bounds__(256) f32_to_bf16_kernel(
    const float* __restrict__ x, __bf16* __restrict__ y) {
  size_t i = ((size_t)blockIdx.x * blockDim.x + threadIdx.x) * 4;
  float4 v = *(const float4*)(x + i);
  bf16x4 o;
  o.x = (__bf16)v.x; o.y = (__bf16)v.y; o.z = (__bf16)v.z; o.w = (__bf16)v.w;
  *(bf16x4*)(y + i) = o;
}

// ---------------------------------------------------------------------------
// Transpose fp32 [K][N] -> bf16 [N][K] via 32x32 LDS tile (block 32x8)
// ---------------------------------------------------------------------------
__global__ void __launch_bounds__(256) transpose_f32_to_bf16(
    const float* __restrict__ W, __bf16* __restrict__ Wt, int Kd, int Nd) {
  __shared__ float tile[32][33];
  int n0 = blockIdx.x * 32;
  int k0 = blockIdx.y * 32;
  int tx = threadIdx.x;   // 0..31
  int ty = threadIdx.y;   // 0..7
#pragma unroll
  for (int rr = 0; rr < 4; ++rr) {
    int r = ty + rr * 8;
    tile[r][tx] = W[(size_t)(k0 + r) * Nd + n0 + tx];
  }
  __syncthreads();
#pragma unroll
  for (int rr = 0; rr < 4; ++rr) {
    int r = ty + rr * 8;
    Wt[(size_t)(n0 + r) * Kd + k0 + tx] = (__bf16)tile[tx][r];
  }
}

// ---------------------------------------------------------------------------
// TDM-staged bf16 GEMM: C[M][N] (fp32) = A[M][K] * Bt[N][K]^T.
// Block = 256 threads (8 waves), 128x128 output tile, K staged in 32-wide
// slabs via tensor_load_to_lds into double-buffered padded LDS tiles.
// Wave 0 drives the TDM pipeline (TENSORcnt); all waves compute from LDS.
// ---------------------------------------------------------------------------
__global__ void __launch_bounds__(256) gemm_bf16_tdm_kernel(
    const __bf16* __restrict__ A, const __bf16* __restrict__ Bt,
    float* __restrict__ C, int M, int N, int Kd) {
  __shared__ __align__(16) __bf16 Al[2][128 * PITCH];   // 2 x 10 KB
  __shared__ __align__(16) __bf16 Bl[2][128 * PITCH];   // 2 x 10 KB

  int m0 = blockIdx.y * 128;
  int n0 = blockIdx.x * 128;
  int w    = threadIdx.x >> 5;       // 0..7
  int lane = threadIdx.x & 31;
  int mr   = lane & 15;
  int half = lane >> 4;
  bool issuer = (w == 0);            // wave-uniform branch (EXEC stays full)

  int niter = Kd / KSTEP;

  if (issuer) {
    tdm_load_tile(lds_off(&Al[0][0]), A  + (size_t)m0 * Kd, Kd, Kd, M);
    tdm_load_tile(lds_off(&Bl[0][0]), Bt + (size_t)n0 * Kd, Kd, Kd, N);
  }

  f32x8 acc[8];
#pragma unroll
  for (int j = 0; j < 8; ++j) acc[j] = {};

  for (int it = 0; it < niter; ++it) {
    int cur = it & 1;
    if (issuer) {
      if (it + 1 < niter) {
        int nxt = cur ^ 1;
        const __bf16* ga = A  + (size_t)m0 * Kd + (size_t)(it + 1) * KSTEP;
        const __bf16* gb = Bt + (size_t)n0 * Kd + (size_t)(it + 1) * KSTEP;
        tdm_load_tile(lds_off(&Al[nxt][0]), ga, Kd, Kd, M);
        tdm_load_tile(lds_off(&Bl[nxt][0]), gb, Kd, Kd, N);
        // per-wave TDM is in-order: <=2 outstanding means tile `it` is resident
        __builtin_amdgcn_s_wait_tensorcnt((short)2);
      } else {
        __builtin_amdgcn_s_wait_tensorcnt((short)0);
      }
    }
    __syncthreads();

    const __bf16* arow  = &Al[cur][(16 * w + mr) * PITCH];
    bf16x16 a = load_frag(arow, half * 8, 16 + half * 8);
    const __bf16* bbase = &Bl[cur][mr * PITCH];
#pragma unroll
    for (int j = 0; j < 8; ++j) {
      bf16x16 b = load_frag(bbase + j * 16 * PITCH, half * 8, 16 + half * 8);
      acc[j] = wmma_bf16(a, b, acc[j]);
    }
    __syncthreads();   // protect buffer reuse before next TDM issue
  }

  int row0 = m0 + 16 * w;
#pragma unroll
  for (int j = 0; j < 8; ++j)
#pragma unroll
    for (int r = 0; r < 8; ++r)
      C[(size_t)(row0 + r + 8 * half) * N + n0 + j * 16 + mr] = acc[j][r];
}

// ---------------------------------------------------------------------------
// Per-head RMSNorm + neox RoPE (+ D^-1/2 folded into Q). Writes bf16:
//   Q[t][h][d], K[t][kvh][d], Vt[kvh][d][t] (transposed for PV fragments)
// grid = (T, NH + 2*NKV), block = 128 (= head_dim)
// ---------------------------------------------------------------------------
__global__ void __launch_bounds__(128) normrope_kernel(
    const float* __restrict__ qkv, const int* __restrict__ positions,
    const float* __restrict__ qw, const float* __restrict__ kw,
    __bf16* __restrict__ Q, __bf16* __restrict__ K, __bf16* __restrict__ Vt) {
  int t  = blockIdx.x;
  int hh = blockIdx.y;          // 0..15 q, 16..23 k, 24..31 v (uniform per block)
  int d  = threadIdx.x;         // 0..127
  const float* row = qkv + (size_t)t * QKV_N;

  if (hh >= NH + NKV) {         // V: plain convert, transposed layout
    int h = hh - (NH + NKV);
    float x = row[(NH + NKV) * HD + h * HD + d];
    Vt[(size_t)(h * HD + d) * T_TOK + t] = (__bf16)x;
    return;
  }

  bool isq = hh < NH;
  int h = isq ? hh : hh - NH;
  float x = isq ? row[h * HD + d] : row[NH * HD + h * HD + d];

  __shared__ float red[128];
  __shared__ float sh[128];
  red[d] = x * x;
  __syncthreads();
#pragma unroll
  for (int s = 64; s > 0; s >>= 1) {
    if (d < s) red[d] += red[d + s];
    __syncthreads();
  }
  float rr = rsqrtf(red[0] * (1.0f / 128.0f) + 1e-6f);
  float xn = x * rr * (isq ? qw[d] : kw[d]);
  sh[d] = xn;
  __syncthreads();

  int i = d & 63;
  float inv = __expf(-(float)i * 0.14391156831212787f);   // 10000^(-2i/128)
  float f = (float)positions[t] * inv;
  float c = __cosf(f), s_ = __sinf(f);
  float other = sh[d ^ 64];
  float out = (d < 64) ? (xn * c - other * s_) : (xn * c + other * s_);

  if (isq) {
    out *= 0.08838834764831845f;          // 128^-0.5 folded into Q
    Q[((size_t)t * NH + h) * HD + d] = (__bf16)out;
  } else {
    K[((size_t)t * NKV + h) * HD + d] = (__bf16)out;
  }
}

// ---------------------------------------------------------------------------
// Causal flash attention. One wave = (head, 16-query tile); streams 32-key
// blocks. QK^T and PV both on v_wmma_f32_16x16x32_bf16; P re-layout via LDS.
// grid = NH*(T/16)/4 blocks of 128 threads (4 waves).
// ---------------------------------------------------------------------------
__global__ void __launch_bounds__(128) attn_kernel(
    const __bf16* __restrict__ Q, const __bf16* __restrict__ Kb,
    const __bf16* __restrict__ Vt, __bf16* __restrict__ O) {
  __shared__ __align__(16) __bf16 pbuf[4][16][32];

  int wave   = (int)((blockIdx.x * blockDim.x + threadIdx.x) >> 5);
  int lane   = threadIdx.x & 31;
  int wlocal = (threadIdx.x >> 5) & 3;
  int h   = wave & (NH - 1);
  int qt  = wave >> 4;
  int q0  = qt * 16;
  int kvh = h >> 1;             // GQA group = 2
  int mr  = lane & 15;
  int half = lane >> 4;

  // Q A-fragments for the whole D=128 (4 k-steps), loaded once.
  const __bf16* qRow = Q + ((size_t)(q0 + mr) * NH + h) * HD;
  bf16x16 qa[4];
#pragma unroll
  for (int ks = 0; ks < 4; ++ks)
    qa[ks] = load_frag(qRow, ks * 32 + half * 8, ks * 32 + 16 + half * 8);

  f32x8 acc[8];
#pragma unroll
  for (int c = 0; c < 8; ++c) acc[c] = {};
  float mstate[8], lstate[8];
#pragma unroll
  for (int r = 0; r < 8; ++r) { mstate[r] = -__builtin_inff(); lstate[r] = 0.f; }

  int kend = q0 + 15;
  for (int kb = 0; kb <= kend; kb += 32) {
    // ---- scores: two 16x16 tiles over this 32-key block ----
    f32x8 s0 = {}, s1 = {};
    const __bf16* kcol0 = Kb + ((size_t)(kb      + mr) * NKV + kvh) * HD;
    const __bf16* kcol1 = Kb + ((size_t)(kb + 16 + mr) * NKV + kvh) * HD;
#pragma unroll
    for (int ks = 0; ks < 4; ++ks) {
      int o0 = ks * 32 + half * 8;
      int o1 = ks * 32 + 16 + half * 8;
      s0 = wmma_bf16(qa[ks], load_frag(kcol0, o0, o1), s0);
      s1 = wmma_bf16(qa[ks], load_frag(kcol1, o0, o1), s1);
    }

    // ---- causal mask + online softmax (rows live in a 16-lane half) ----
    float fac[8];
#pragma unroll
    for (int r = 0; r < 8; ++r) {
      int m = q0 + r + 8 * half;
      if (kb + mr > m)      s0[r] = -__builtin_inff();
      if (kb + 16 + mr > m) s1[r] = -__builtin_inff();
      float mx = fmaxf(s0[r], s1[r]);
      mx = fmaxf(mx, __shfl_xor(mx, 1, 32));
      mx = fmaxf(mx, __shfl_xor(mx, 2, 32));
      mx = fmaxf(mx, __shfl_xor(mx, 4, 32));
      mx = fmaxf(mx, __shfl_xor(mx, 8, 32));
      float mnew = fmaxf(mstate[r], mx);
      fac[r] = __expf(mstate[r] - mnew);
      float p0 = __expf(s0[r] - mnew);
      float p1 = __expf(s1[r] - mnew);
      float sum = p0 + p1;
      sum += __shfl_xor(sum, 1, 32);
      sum += __shfl_xor(sum, 2, 32);
      sum += __shfl_xor(sum, 4, 32);
      sum += __shfl_xor(sum, 8, 32);
      lstate[r] = lstate[r] * fac[r] + sum;
      mstate[r] = mnew;
      // stage P (C-layout -> LDS [m][k]) for the PV A-fragment
      pbuf[wlocal][r + 8 * half][mr]      = (__bf16)p0;
      pbuf[wlocal][r + 8 * half][16 + mr] = (__bf16)p1;
    }
#pragma unroll
    for (int c = 0; c < 8; ++c)
#pragma unroll
      for (int r = 0; r < 8; ++r) acc[c][r] *= fac[r];

    // same-wave DS ordering; explicit split-counter wait before re-read
    asm volatile("s_wait_dscnt 0" ::: "memory");
    bf16x16 pa = load_frag(&pbuf[wlocal][mr][0], half * 8, 16 + half * 8);

    // ---- PV: out[16x128] += P[16x32] @ V[32x128] ----
    const __bf16* vcol = Vt + (size_t)(kvh * HD + mr) * T_TOK + kb;
#pragma unroll
    for (int c = 0; c < 8; ++c) {
      bf16x16 bv = load_frag(vcol + (size_t)c * 16 * T_TOK, half * 8,
                             16 + half * 8);
      acc[c] = wmma_bf16(pa, bv, acc[c]);
    }
  }

#pragma unroll
  for (int c = 0; c < 8; ++c)
#pragma unroll
    for (int r = 0; r < 8; ++r) {
      float o = acc[c][r] / lstate[r];
      O[(size_t)(q0 + r + 8 * half) * (NH * HD) + h * HD + c * 16 + mr] =
          (__bf16)o;
    }
}

// ---------------------------------------------------------------------------
// Host-side orchestration
// ---------------------------------------------------------------------------
extern "C" void kernel_launch(void* const* d_in, const int* in_sizes, int n_in,
                              void* d_out, int out_size, void* d_ws,
                              size_t ws_size, hipStream_t stream) {
  (void)in_sizes; (void)n_in; (void)out_size; (void)ws_size;
  const int*   positions = (const int*)  d_in[0];
  const float* hidden    = (const float*)d_in[1];
  const float* w_qkv     = (const float*)d_in[2];
  const float* w_o       = (const float*)d_in[3];
  const float* q_norm_w  = (const float*)d_in[4];
  const float* k_norm_w  = (const float*)d_in[5];
  float* out = (float*)d_out;

  char* ws = (char*)d_ws;
  size_t off = 0;
  __bf16* hb    = (__bf16*)(ws + off); off += (size_t)T_TOK * HIDN  * 2;  // hidden bf16
  __bf16* wqkvT = (__bf16*)(ws + off); off += (size_t)QKV_N * HIDN  * 2;  // [N][K]
  __bf16* woT   = (__bf16*)(ws + off); off += (size_t)HIDN  * HIDN  * 2;  // [N][K]
  float*  qkv   = (float*) (ws + off); off += (size_t)T_TOK * QKV_N * 4;  // fp32
  __bf16* Qb    = (__bf16*)(ws + off); off += (size_t)T_TOK * NH  * HD * 2;
  __bf16* Kb    = (__bf16*)(ws + off); off += (size_t)T_TOK * NKV * HD * 2;
  __bf16* Vt    = (__bf16*)(ws + off); off += (size_t)NKV * HD * T_TOK * 2;
  __bf16* Ob    = (__bf16*)(ws + off); off += (size_t)T_TOK * NH  * HD * 2;

  // 1) hidden fp32 -> bf16
  f32_to_bf16_kernel<<<(T_TOK * HIDN / 4) / 256, 256, 0, stream>>>(hidden, hb);

  // 2) transpose+convert weights to [N][K] bf16
  transpose_f32_to_bf16<<<dim3(QKV_N / 32, HIDN / 32), dim3(32, 8), 0, stream>>>(
      w_qkv, wqkvT, HIDN, QKV_N);
  transpose_f32_to_bf16<<<dim3(HIDN / 32, HIDN / 32), dim3(32, 8), 0, stream>>>(
      w_o, woT, HIDN, HIDN);

  // 3) QKV projection: [T,HID] x [HID,QKV_N] -> fp32 [T,QKV_N]  (TDM-staged)
  gemm_bf16_tdm_kernel<<<dim3(QKV_N / 128, T_TOK / 128), 256, 0, stream>>>(
      hb, wqkvT, qkv, T_TOK, QKV_N, HIDN);

  // 4) per-head RMSNorm + RoPE, split into Q/K/Vt bf16
  normrope_kernel<<<dim3(T_TOK, NH + 2 * NKV), 128, 0, stream>>>(
      qkv, positions, q_norm_w, k_norm_w, Qb, Kb, Vt);

  // 5) causal flash attention -> Ob bf16 [T, NH*HD]
  {
    int waves = NH * (T_TOK / 16);
    attn_kernel<<<waves * 32 / 128, 128, 0, stream>>>(Qb, Kb, Vt, Ob);
  }

  // 6) output projection: [T, NH*HD] x [NH*HD, HID] -> d_out fp32 (TDM-staged)
  gemm_bf16_tdm_kernel<<<dim3(HIDN / 128, T_TOK / 128), 256, 0, stream>>>(
      Ob, woT, out, T_TOK, HIDN, NH * HD);
}
